// BaseLineModel_57973468562070
// MI455X (gfx1250) — compile-verified, matrix operand order checked by reading, not run on hardware
//
#include <hip/hip_runtime.h>
#include <hip/hip_bf16.h>

// Dims from the reference
#define NN 16384   // notes
#define LL 64      // tokens per note
#define EE 64      // embedding dim
#define HH 256     // conv output channels
#define KK 3       // conv kernel width
#define SS 1024    // segments (days)
#define NPW 4            // notes per workgroup
#define ROWPITCH 72      // LDS row pitch in bf16 elems (64 data + 8 pad)
#define AROWS 66         // 64 token rows + 2 zero pad rows (t+tap <= 65)
#define NFRAG (6*16)     // 6 K-chunks x 16 H-tiles = 96 B fragments
#define SUMROW 258       // [deltaSum, 256 featSums, count]

typedef __attribute__((ext_vector_type(16))) __bf16 v16bf;
typedef __attribute__((ext_vector_type(8)))  float  v8f;

union Frag16 { v16bf v; uint4 q[2]; };

__device__ __forceinline__ unsigned short f2bf(float f) {
    unsigned u = __float_as_uint(f);
    return (unsigned short)((u + 0x7FFFu + ((u >> 16) & 1u)) >> 16);  // RNE
}

// --- Kernel 0: rearrange conv_w [H][E][K] f32 -> bf16 B-fragments in d_ws ---
// Fragment layout: bfrag[((kc*16+nt)*32 + lane)*16 + j], j per ISA 16-bit B layout:
// klocal = (j<8 ? j : 16+(j-8)) + 8*(lane>>4);  c = kc*32+klocal; tap=c/64; e=c%64;
// h = nt*16 + (lane&15);  value = conv_w[h][e][tap]
__global__ __launch_bounds__(256) void wprep_kernel(const float* __restrict__ conv_w,
                                                    unsigned short* __restrict__ bfrag) {
    int idx = blockIdx.x * 256 + threadIdx.x;
    if (idx >= NFRAG * 32 * 16) return;
    int j    = idx & 15;
    int lane = (idx >> 4) & 31;
    int fi   = idx >> 9;          // 0..95
    int kc   = fi >> 4;
    int nt   = fi & 15;
    int klocal = ((j < 8) ? j : (16 + (j - 8))) + ((lane >> 4) << 3);
    int c   = kc * 32 + klocal;
    int tap = c >> 6;
    int e   = c & 63;
    int h   = nt * 16 + (lane & 15);
    bfrag[idx] = f2bf(conv_w[(h * EE + e) * KK + tap]);
}

// --- Kernel 1: zero the segment accumulator ---
__global__ __launch_bounds__(256) void zero_kernel(float* __restrict__ sums, int n) {
    int i = blockIdx.x * 256 + threadIdx.x;
    if (i < n) sums[i] = 0.0f;
}

// --- Kernel 2: embed -> conv (WMMA bf16) -> relu -> maxpool -> segment atomic add ---
__global__ __launch_bounds__(256, 2) void conv_kernel(const int* __restrict__ text,
                                                      const int* __restrict__ st,
                                                      const float* __restrict__ emb,
                                                      const float* __restrict__ conv_b,
                                                      const unsigned short* __restrict__ bfrag,
                                                      float* __restrict__ sums) {
    __shared__ unsigned short sA[NPW * AROWS * ROWPITCH];   // 38016 B bf16 tiles

    const int tid = threadIdx.x;

    // ---- Stage A: gather 4 notes x 64 token embedding rows into LDS as bf16 ----
    {
        int note = tid >> 6;            // 0..3
        int t    = tid & 63;            // token row
        int tok  = text[(blockIdx.x * NPW + note) * LL + t];
        const float4* er = (const float4*)(emb + (size_t)tok * EE);
        unsigned short* dst = sA + note * (AROWS * ROWPITCH) + t * ROWPITCH;
#pragma unroll
        for (int i = 0; i < 16; ++i) {
            float4 f = er[i];
            dst[i * 4 + 0] = f2bf(f.x);
            dst[i * 4 + 1] = f2bf(f.y);
            dst[i * 4 + 2] = f2bf(f.z);
            dst[i * 4 + 3] = f2bf(f.w);
        }
    }
    // zero the two pad rows (rows 64,65) of each note tile: 4*2*64 = 512 elems
    if (tid < 128) {
        int idx  = tid * 4;             // 0..508
        int note = idx >> 7;
        int rem  = idx & 127;
        int r    = 64 + (rem >> 6);
        int e    = rem & 63;
        unsigned short* dst = sA + note * (AROWS * ROWPITCH) + r * ROWPITCH + e;
#pragma unroll
        for (int k = 0; k < 4; ++k) dst[k] = 0;
    }
    __syncthreads();

    // ---- Per-wave GEMM: 2 waves per note, 8 H-tiles per wave ----
    const int wave   = tid >> 5;
    const int lane   = tid & 31;
    const int note   = wave >> 1;            // 0..3
    const int ntBase = (wave & 1) * 8;       // H-tile range [ntBase, ntBase+8)
    const int half   = lane >> 4;            // lane half
    const int mrow   = lane & 15;            // A-matrix row within tile
    const int nglob  = blockIdx.x * NPW + note;
    const unsigned short* A = sA + note * (AROWS * ROWPITCH);

    // hoist per-channel bias into registers (one per H-tile handled by this wave)
    float cbv[8];
#pragma unroll
    for (int j = 0; j < 8; ++j) cbv[j] = conv_b[(ntBase + j) * 16 + (lane & 15)];

    float featmax[8];
#pragma unroll
    for (int j = 0; j < 8; ++j) featmax[j] = 0.0f;   // relu >= 0, so 0 is identity

#pragma unroll 1
    for (int mt = 0; mt < 4; ++mt) {
        // Load the 6 A fragments for this 16-row M tile (reused across 8 N tiles).
        Frag16 a[6];
        const int t0 = mt * 16 + mrow;
#pragma unroll
        for (int kc = 0; kc < 6; ++kc) {
            int tap = kc >> 1;
            int eb  = (kc & 1) * 32;
            const unsigned short* rowp = A + (t0 + tap) * ROWPITCH + eb + half * 8;
            a[kc].q[0] = *(const uint4*)(rowp);
            a[kc].q[1] = *(const uint4*)(rowp + 16);
        }
#pragma unroll
        for (int j = 0; j < 8; ++j) {
            const int nt = ntBase + j;
            v8f acc = {};
#pragma unroll
            for (int kc = 0; kc < 6; ++kc) {
                Frag16 bf;
                const uint4* bp = (const uint4*)(bfrag + ((size_t)((kc * 16 + nt) * 32 + lane)) * 16);
                bf.q[0] = bp[0];
                bf.q[1] = bp[1];
                acc = __builtin_amdgcn_wmma_f32_16x16x32_bf16(
                        false, a[kc].v, false, bf.v, (short)0, acc, false, false);
            }
            // bias + relu + mask t>=62, then max over the 8 accumulator rows
            float p = 0.0f;
#pragma unroll
            for (int r = 0; r < 8; ++r) {
                int t = mt * 16 + r + 8 * half;   // C layout: M = r + 8*(lane/16)
                float v = acc[r] + cbv[j];
                v = v > 0.0f ? v : 0.0f;
                p = fmaxf(p, (t < (LL - KK + 1)) ? v : 0.0f);
            }
            // fold the two lane halves (same N = lane%16)
            p = fmaxf(p, __shfl_xor(p, 16, 32));
            featmax[j] = fmaxf(featmax[j], p);
        }
    }

    // ---- Segment accumulation ----
    const int s = st[nglob];
    float* srow = sums + (size_t)s * SUMROW;
    if (lane < 16) {
#pragma unroll
        for (int j = 0; j < 8; ++j) {
            int h = (ntBase + j) * 16 + lane;
            atomicAdd(srow + 1 + h, featmax[j]);
        }
    }
    // one lane per note handles delta + count
    if ((wave & 1) == 0 && lane == 16) {
        float delta = (nglob == 0) ? 0.0f : (float)(s - st[nglob - 1]);
        atomicAdd(srow + 0, delta);
        atomicAdd(srow + SUMROW - 1, 1.0f);
    }
}

// --- Kernel 3: segment mean -> linear(257->1) -> sigmoid ---
__global__ __launch_bounds__(256) void final_kernel(const float* __restrict__ sums,
                                                    const float* __restrict__ W,
                                                    const float* __restrict__ b,
                                                    float* __restrict__ out) {
    int s = blockIdx.x * 256 + threadIdx.x;
    if (s >= SS) return;
    const float* row = sums + (size_t)s * SUMROW;
    float cnt = row[SUMROW - 1];
    float inv = 1.0f / fmaxf(cnt, 1.0f);
    float acc = b[0];
#pragma unroll 1
    for (int j = 0; j < HH + 1; ++j) acc += row[j] * inv * W[j];
    out[s] = 1.0f / (1.0f + __expf(-acc));
}

extern "C" void kernel_launch(void* const* d_in, const int* in_sizes, int n_in,
                              void* d_out, int out_size, void* d_ws, size_t ws_size,
                              hipStream_t stream) {
    const int*   text   = (const int*)d_in[0];
    const int*   st     = (const int*)d_in[1];
    const float* emb    = (const float*)d_in[2];
    const float* conv_w = (const float*)d_in[3];
    const float* conv_b = (const float*)d_in[4];
    const float* W      = (const float*)d_in[5];
    const float* b      = (const float*)d_in[6];
    float* out = (float*)d_out;

    // Workspace layout: [0, 96KB) bf16 B fragments; then S*258 f32 accumulator.
    unsigned short* bfrag = (unsigned short*)d_ws;
    float* sums = (float*)((char*)d_ws + (size_t)NFRAG * 32 * 16 * sizeof(unsigned short));

    // 0) pack conv weights into WMMA B-fragment order (bf16)
    wprep_kernel<<<(NFRAG * 32 * 16 + 255) / 256, 256, 0, stream>>>(conv_w, bfrag);
    // 1) zero segment accumulator
    const int nsum = SS * SUMROW;
    zero_kernel<<<(nsum + 255) / 256, 256, 0, stream>>>(sums, nsum);
    // 2) main conv + pool + segment sum
    conv_kernel<<<NN / NPW, 256, 0, stream>>>(text, st, emb, conv_b, bfrag, sums);
    // 3) mean + linear + sigmoid
    final_kernel<<<(SS + 255) / 256, 256, 0, stream>>>(sums, W, b, out);
}
